// PointerNet_42245298323491
// MI455X (gfx1250) — matrix operand order repaired and checked by reading, not compiled, not run on hardware
//
#include <hip/hip_runtime.h>

// Problem constants
#define SEQ   50
#define HID   256
#define G4H   1024            // 4*HID
#define NPC   3200            // sequences per call (B*X)
#define BLKS  200             // NPC / 16
#define ROWSB 800             // 16 * SEQ rows of enc per block

typedef unsigned short bhalf;     // raw bf16 bits
typedef unsigned int   u32_t;

typedef __attribute__((ext_vector_type(16))) __bf16 v16bf;
typedef __attribute__((ext_vector_type(8)))  float  v8f;

union FragB { v16bf v; bhalf u[16]; uint4 q[2]; };
union FragD { v8f v; float f[8]; float4 q[2]; };

#if __has_builtin(__builtin_amdgcn_tanhf)
#define TANHF(x) __builtin_amdgcn_tanhf(x)
#else
#define TANHF(x) tanhf(x)
#endif

#if __has_builtin(__builtin_amdgcn_rcpf)
#define RCPF(x) __builtin_amdgcn_rcpf(x)
#else
#define RCPF(x) (1.0f / (x))
#endif

__device__ __forceinline__ float b2f(bhalf b) {
    union { u32_t u; float f; } c; c.u = ((u32_t)b) << 16; return c.f;
}
// native f32 -> bf16 (hardware cvt, pairs fuse into v_cvt_pk_bf16_f32)
__device__ __forceinline__ bhalf f2b(float x) {
    union { __bf16 h; bhalf u; } c; c.h = (__bf16)x; return c.u;
}
__device__ __forceinline__ float sigm(float x) {
    return RCPF(1.f + __expf(-x));
}

__device__ __forceinline__ v8f wmma_bf(v16bf a, v16bf b, v8f c) {
    // D = A(16x32 bf16) * B(32x16 bf16) + C(16x16 f32)
    return __builtin_amdgcn_wmma_f32_16x16x32_bf16(false, a, false, b,
                                                   (short)0, c, false, false);
}

// A fragment (16x32 bf16) from an LDS tile laid out [16][HID] (bf16 bits).
// ISA layout: lanes0-15 row M=lane, elems0..7 = K k0+8*hi.. ; elems8..15 = K k0+16+8*hi..
__device__ __forceinline__ v16bf ldsA(const bhalf* t, int k0, int lane) {
    int m = lane & 15, hi = (lane >> 4) & 1;
    const bhalf* r = t + m * HID + k0 + 8 * hi;
    FragB f;
    f.q[0] = *(const uint4*)(r);
    f.q[1] = *(const uint4*)(r + 16);
    return f.v;
}

// B fragment (32x16 bf16) = W^T tile; W is row-major [J][HID] bf16 in global.
__device__ __forceinline__ v16bf glbB(const bhalf* __restrict__ W, int j0, int k0, int lane) {
    const bhalf* p = W + (size_t)(j0 + (lane & 15)) * HID + k0 + ((lane >> 4) << 3);
    FragB f;
    f.q[0] = *(const uint4*)(p);
    f.q[1] = *(const uint4*)(p + 16);
    return f.v;
}

// A fragment from global fp32 rows (row-tile rt of the per-block 800x256 slab)
__device__ __forceinline__ v16bf glbAf32(const float* __restrict__ slab, int rt, int k0, int lane) {
    int m = lane & 15, hi = (lane >> 4) & 1;
    const float* r = slab + (size_t)(rt * 16 + m) * HID + k0 + 8 * hi;
    float4 a0 = *(const float4*)(r);
    float4 a1 = *(const float4*)(r + 4);
    float4 b0 = *(const float4*)(r + 16);
    float4 b1 = *(const float4*)(r + 20);
    FragB f;
    f.u[0] = f2b(a0.x); f.u[1] = f2b(a0.y); f.u[2]  = f2b(a0.z); f.u[3]  = f2b(a0.w);
    f.u[4] = f2b(a1.x); f.u[5] = f2b(a1.y); f.u[6]  = f2b(a1.z); f.u[7]  = f2b(a1.w);
    f.u[8] = f2b(b0.x); f.u[9] = f2b(b0.y); f.u[10] = f2b(b0.z); f.u[11] = f2b(b0.w);
    f.u[12] = f2b(b1.x); f.u[13] = f2b(b1.y); f.u[14] = f2b(b1.z); f.u[15] = f2b(b1.w);
    return f.v;
}

// Store D fragment transposed into LDS T[j][m] (row length 16 floats).
// VGPR r: lanes0-15 -> M=r, lanes16-31 -> M=r+8; N = lane&15.
__device__ __forceinline__ void stDT(float* T, int jt, v8f d, int lane) {
    int j = jt * 16 + (lane & 15);
    int hi = (lane >> 4) & 1;
    FragD f; f.v = d;
    *(float4*)(T + j * 16 + hi * 8)     = f.q[0];
    *(float4*)(T + j * 16 + hi * 8 + 4) = f.q[1];
}

// ---------------------------------------------------------------------------
// Kernel 1: convert weights to bf16, fuse biases
// ---------------------------------------------------------------------------
__global__ __launch_bounds__(256) void pn_convert(
    const float* __restrict__ Wih, const float* __restrict__ Whh,
    const float* __restrict__ W1,  const float* __restrict__ W2,
    const float* __restrict__ bih, const float* __restrict__ bhh,
    bhalf* __restrict__ WihB, bhalf* __restrict__ WhhB,
    bhalf* __restrict__ W1B,  bhalf* __restrict__ W2B,
    float* __restrict__ biasS)
{
    int i = blockIdx.x * 256 + threadIdx.x;   // grid = 1024*256 = 262144 threads
    if (i < G4H * HID) { WihB[i] = f2b(Wih[i]); WhhB[i] = f2b(Whh[i]); }
    if (i < HID * HID) { W1B[i]  = f2b(W1[i]);  W2B[i]  = f2b(W2[i]);  }
    if (i < G4H)       { biasS[i] = bih[i] + bhh[i]; }
}

// ---------------------------------------------------------------------------
// Kernel 2: full pointer-net decode. One block = 16 sequences, 8 wave32s.
// ---------------------------------------------------------------------------
__global__ __launch_bounds__(256) void pn_decode(
    const float* __restrict__ enc0, const float* __restrict__ enc1,
    const float* __restrict__ enc2,
    const bhalf* __restrict__ Wih, const bhalf* __restrict__ Whh,
    const bhalf* __restrict__ W2b, const bhalf* __restrict__ W1b,
    const float* __restrict__ biasS, const float* __restrict__ vt,
    bhalf* __restrict__ enctAll, float* __restrict__ out)
{
    const int tid  = threadIdx.x;
    const int lane = tid & 31;
    const int wave = tid >> 5;
    const int bx   = blockIdx.x;          // 0..199
    const int call = blockIdx.y;          // 0..2
    const float* enc = (call == 0) ? enc0 : (call == 1) ? enc1 : enc2;
    const int n0 = bx * 16;

    bhalf* enct = enctAll + (size_t)(call * BLKS + bx) * (ROWSB * HID);
    float* outS = out + (size_t)call * (NPC * SEQ * SEQ + NPC * SEQ);
    float* outI = outS + (size_t)NPC * SEQ * SEQ;

    // ---- dynamic LDS carve (123328 bytes total) ----
    extern __shared__ char smem[];
    bhalf* xq     = (bhalf*)(smem);               // [16][256] bf16   (8192)
    bhalf* hq     = (bhalf*)(smem + 8192);        // [16][256] bf16   (8192)
    float* gatesT = (float*)(smem + 16384);       // [1024][16] f32   (65536)
    float* decT   = (float*)(smem + 81920);       // [256][16]  f32   (16384)
    float* cst    = (float*)(smem + 98304);       // [16][256]  f32   (16384)
    float* uL     = (float*)(smem + 114688);      // [16][52]   f32   (3328)
    float* vtL    = (float*)(smem + 118016);      // [256]            (1024)
    float* biasL  = (float*)(smem + 119040);      // [1024]           (4096)
    float* smx    = (float*)(smem + 123136);      // [16]
    float* sinv   = (float*)(smem + 123200);      // [16]
    int*   sidx   = (int*)  (smem + 123264);      // [16]

    // ---- phase 0a: vt + bias into LDS, x = 0, h0 = c0 = sum_s enc ----
    vtL[tid] = vt[tid];
    for (int j = tid; j < G4H; j += 256) biasL[j] = biasS[j];

    {
        const int m = tid >> 4, c0 = (tid & 15) * 16;
        const float* base = enc + ((size_t)(n0 + m) * SEQ) * HID + c0;
        float4 s0 = {0.f,0.f,0.f,0.f}, s1 = s0, s2 = s0, s3 = s0;
        for (int s = 0; s < SEQ; ++s) {
            const float4* p4 = (const float4*)(base + (size_t)s * HID);
            float4 a = p4[0], b = p4[1], c = p4[2], d = p4[3];
            s0.x += a.x; s0.y += a.y; s0.z += a.z; s0.w += a.w;
            s1.x += b.x; s1.y += b.y; s1.z += b.z; s1.w += b.w;
            s2.x += c.x; s2.y += c.y; s2.z += c.z; s2.w += c.w;
            s3.x += d.x; s3.y += d.y; s3.z += d.z; s3.w += d.w;
        }
        float acc[16] = { s0.x,s0.y,s0.z,s0.w, s1.x,s1.y,s1.z,s1.w,
                          s2.x,s2.y,s2.z,s2.w, s3.x,s3.y,s3.z,s3.w };
#pragma unroll
        for (int i = 0; i < 16; ++i) {
            cst[m * HID + c0 + i] = acc[i];
            hq [m * HID + c0 + i] = f2b(acc[i]);
            xq [m * HID + c0 + i] = 0;
        }
    }

    // ---- phase 0b: enc_t = enc @ W1^T  (bf16 WMMA, per-block slab -> ws) ----
    {
        const float* slab = enc + (size_t)n0 * SEQ * HID;   // 800 x 256 rows
        const int jt0 = wave * 2, jt1 = wave * 2 + 1;       // 16 col-tiles / 8 waves
#pragma unroll 1
        for (int rt = 0; rt < SEQ; ++rt) {                  // 50 row-tiles
            v8f acc0 = {}, acc1 = {};
#pragma unroll 1
            for (int kb = 0; kb < 8; ++kb) {
                int k0 = kb * 32;
                v16bf A = glbAf32(slab, rt, k0, lane);
                acc0 = wmma_bf(A, glbB(W1b, jt0 * 16, k0, lane), acc0);
                acc1 = wmma_bf(A, glbB(W1b, jt1 * 16, k0, lane), acc1);
            }
            int hi = (lane >> 4) & 1;
            FragD f0; f0.v = acc0;
            FragD f1; f1.v = acc1;
            int ja = jt0 * 16 + (lane & 15);
            int jb = jt1 * 16 + (lane & 15);
#pragma unroll
            for (int r = 0; r < 8; ++r) {
                int row = rt * 16 + r + 8 * hi;
                enct[(size_t)row * HID + ja] = f2b(f0.f[r]);
                enct[(size_t)row * HID + jb] = f2b(f1.f[r]);
            }
        }
    }
    __threadfence();
    __syncthreads();

    // ---- 50 decode steps ----
#pragma unroll 1
    for (int step = 0; step < SEQ; ++step) {
        // 1) gates = x@Wih^T + h@Whh^T  -> gatesT[1024][16]
        //    two passes of 4 col-tiles to keep VGPR pressure < 256
#pragma unroll 1
        for (int half = 0; half < 2; ++half) {
            v8f acc[4];
#pragma unroll
            for (int i = 0; i < 4; ++i) acc[i] = (v8f){};
#pragma unroll 1
            for (int kb = 0; kb < 8; ++kb) {
                int k0 = kb * 32;
                v16bf ax = ldsA(xq, k0, lane);
                v16bf ah = ldsA(hq, k0, lane);
#pragma unroll
                for (int i = 0; i < 4; ++i) {
                    int j0 = (wave * 8 + half * 4 + i) * 16;
                    acc[i] = wmma_bf(ax, glbB(Wih, j0, k0, lane), acc[i]);
                    acc[i] = wmma_bf(ah, glbB(Whh, j0, k0, lane), acc[i]);
                }
            }
#pragma unroll
            for (int i = 0; i < 4; ++i)
                stDT(gatesT, wave * 8 + half * 4 + i, acc[i], lane);
        }
        __syncthreads();

        // 2) elementwise LSTM cell: c = f*c + i*g ; h = o*tanh(c)
        {
            const int m = tid >> 4, c0 = (tid & 15) * 16;
#pragma unroll
            for (int i = 0; i < 16; ++i) {
                int j = c0 + i;
                float gi = gatesT[(j        ) * 16 + m] + biasL[j];
                float gf = gatesT[(j +  HID ) * 16 + m] + biasL[j + HID];
                float gg = gatesT[(j + 2*HID) * 16 + m] + biasL[j + 2*HID];
                float go = gatesT[(j + 3*HID) * 16 + m] + biasL[j + 3*HID];
                float ig = sigm(gi);
                float fg = sigm(gf);
                float og = sigm(go);
                float tg = TANHF(gg);
                float c  = fg * cst[m * HID + j] + ig * tg;
                cst[m * HID + j] = c;
                hq [m * HID + j] = f2b(og * TANHF(c));
            }
        }
        __syncthreads();

        // 3) dec_t = h @ W2^T -> decT[256][16]
        {
#pragma unroll
            for (int ci = 0; ci < 2; ++ci) {
                int jt = wave * 2 + ci;
                v8f acc = {};
#pragma unroll 1
                for (int kb = 0; kb < 8; ++kb) {
                    int k0 = kb * 32;
                    acc = wmma_bf(ldsA(hq, k0, lane), glbB(W2b, jt * 16, k0, lane), acc);
                }
                stDT(decT, jt, acc, lane);
            }
        }
        __syncthreads();

        // 4) attention: u[m][s] = sum_k vt[k]*tanh(enc_t[m,s,k] + dec_t[m,k])
        //    Each thread owns a fixed m (= tid&15) and rows s0, s0+16, s0+32
        //    (+s0+48 for tid<32), so decT/vt LDS loads amortize over 3-4 rows.
        {
            const int m  = tid & 15;
            const int s0 = tid >> 4;               // 0..15
            const bool has3 = (tid < 32);          // s0+48 valid only then
            const bhalf* e0 = enct + (size_t)(m * SEQ + s0     ) * HID;
            const bhalf* e1 = enct + (size_t)(m * SEQ + s0 + 16) * HID;
            const bhalf* e2 = enct + (size_t)(m * SEQ + s0 + 32) * HID;
            const bhalf* e3 = has3 ? enct + (size_t)(m * SEQ + s0 + 48) * HID : e0;
            float a0 = 0.f, a1 = 0.f, a2 = 0.f, a3 = 0.f;
#pragma unroll 1
            for (int k0 = 0; k0 < HID; k0 += 8) {
                uint4 q0 = *(const uint4*)(e0 + k0);
                uint4 q1 = *(const uint4*)(e1 + k0);
                uint4 q2 = *(const uint4*)(e2 + k0);
                uint4 q3 = *(const uint4*)(e3 + k0);
                const bhalf* p0 = (const bhalf*)&q0;
                const bhalf* p1 = (const bhalf*)&q1;
                const bhalf* p2 = (const bhalf*)&q2;
                const bhalf* p3 = (const bhalf*)&q3;
#pragma unroll
                for (int e = 0; e < 8; ++e) {
                    float d = decT[(k0 + e) * 16 + m];
                    float w = vtL[k0 + e];
                    a0 += w * TANHF(b2f(p0[e]) + d);
                    a1 += w * TANHF(b2f(p1[e]) + d);
                    a2 += w * TANHF(b2f(p2[e]) + d);
                    a3 += w * TANHF(b2f(p3[e]) + d);
                }
            }
            uL[m * 52 + s0]      = a0;
            uL[m * 52 + s0 + 16] = a1;
            uL[m * 52 + s0 + 32] = a2;
            if (has3) uL[m * 52 + s0 + 48] = a3;
        }
        __syncthreads();

        // 5) per-row max / sum / argmax (argmax(softmax) == argmax(u))
        if (tid < 16) {
            int m = tid;
            float mx = -1e30f; int bi = 0;
            for (int s = 0; s < SEQ; ++s) {
                float v = uL[m * 52 + s];
                if (v > mx) { mx = v; bi = s; }
            }
            float sum = 0.f;
            for (int s = 0; s < SEQ; ++s) sum += __expf(uL[m * 52 + s] - mx);
            smx[m] = mx; sinv[m] = RCPF(sum); sidx[m] = bi;
            outI[(size_t)(n0 + m) * SEQ + step] = (float)bi;
        }
        __syncthreads();

        // 6) write softmax scores, gather next x from enc[idx]
        for (int p = tid; p < 16 * SEQ; p += 256) {
            int m = p / SEQ, s = p % SEQ;
            outS[((size_t)(n0 + m) * SEQ + step) * SEQ + s] =
                __expf(uL[m * 52 + s] - smx[m]) * sinv[m];
        }
        {
            const int m = tid >> 4, c0 = (tid & 15) * 16;
            const float* xr = enc + ((size_t)(n0 + m) * SEQ + sidx[m]) * HID + c0;
#pragma unroll
            for (int i = 0; i < 16; ++i) xq[m * HID + c0 + i] = f2b(xr[i]);
        }
        __syncthreads();
    }
}

// ---------------------------------------------------------------------------
extern "C" void kernel_launch(void* const* d_in, const int* in_sizes, int n_in,
                              void* d_out, int out_size, void* d_ws, size_t ws_size,
                              hipStream_t stream)
{
    const float* home = (const float*)d_in[0];
    const float* vis  = (const float*)d_in[1];
    const float* team = (const float*)d_in[2];
    const float* Wih  = (const float*)d_in[3];
    const float* Whh  = (const float*)d_in[4];
    const float* bih  = (const float*)d_in[5];
    const float* bhh  = (const float*)d_in[6];
    const float* W1   = (const float*)d_in[7];
    const float* W2   = (const float*)d_in[8];
    const float* vt   = (const float*)d_in[9];

    char* ws = (char*)d_ws;
    bhalf* WihB = (bhalf*)(ws);                 // 262144 bf16
    bhalf* WhhB = WihB + G4H * HID;             // 262144
    bhalf* W1B  = WhhB + G4H * HID;             // 65536
    bhalf* W2B  = W1B + HID * HID;              // 65536
    float* biasS = (float*)(ws + 1310720);      // 1024 f32
    bhalf* enct  = (bhalf*)(ws + 1314816);      // 3*200*800*256 bf16 (~246 MB)

    pn_convert<<<1024, 256, 0, stream>>>(Wih, Whh, W1, W2, bih, bhh,
                                         WihB, WhhB, W1B, W2B, biasS);

    dim3 grid(BLKS, 3);
    pn_decode<<<grid, 256, 123328, stream>>>(home, vis, team,
                                             WihB, WhhB, W2B, W1B,
                                             biasS, vt, enct, (float*)d_out);
}
